// GraphNetworkBlock_79233556677179
// MI455X (gfx1250) — matrix operand order repaired and checked by reading, not compiled, not run on hardware
//
#include <hip/hip_runtime.h>

// ---------------------------------------------------------------------------
// GraphNetworkBlock on gfx1250 (MI455X).
// Edge MLP (131072x1024 @ 1024x256) dominates: 68.7 GFLOP; memory floor
// ~0.6-0.8 GB -> ~30us @ 23.3 TB/s. GEMMs run on v_wmma_f32_16x16x32_bf16
// (bf16 in, f32 accumulate - keeps the f32 exponent range). B panels are
// staged into LDS with global_load_async_to_lds_b128 (ASYNCcnt path); A
// tiles are gathered (concat segments, optional row indices) + converted
// f32->bf16 through VGPRs. Aggregations are deterministic ordered scans.
// ---------------------------------------------------------------------------

typedef __bf16  bf16_t;
typedef __bf16  v16bf __attribute__((ext_vector_type(16)));
typedef float   v8f   __attribute__((ext_vector_type(8)));

#define DIM 256            // GDIM = NDIM = EDIM = EOUT = NOUT = GOUT
#define APAD 40            // padded LDS row length (elems): 80B rows
#define BPAD 40

static __device__ __forceinline__ bf16_t to_bf16(float f) { return (bf16_t)f; }

// ---------------------------------------------------------------------------
// Weight convert + transpose into K-panel-tiled bf16:
//   Wt[(k/32)][n][k%32]  <-  W[k][n]     (W is [K x 256] f32)
// Each 32-wide K panel of B (256x32 bf16 = 16KB) is contiguous -> the GEMM
// stages it with coalesced async b128 copies straight into LDS.
// ---------------------------------------------------------------------------
__global__ void wconv_kernel(const float* __restrict__ W, bf16_t* __restrict__ Wt,
                             int K) {
    int t = blockIdx.x * blockDim.x + threadIdx.x;
    if (t >= K * DIM) return;
    int kk   = t & 31;
    int rest = t >> 5;          // p*256 + n
    int n    = rest & 255;
    int p    = rest >> 8;
    int k    = (p << 5) + kk;
    Wt[t] = to_bf16(W[(size_t)k * DIM + n]);
}

// ---------------------------------------------------------------------------
// Generic fused gather-GEMM + bias + ReLU (+ optional row mask).
//   out[M x 256] = relu( concat(seg0..seg{nseg-1}) @ W + bias )
// Segments are 256 cols wide; segment s row = idx_s ? idx_s[row] : row.
// Block: 256 threads (8 waves), M-tile = 64, N = 256, K step = 32.
// Wave w: M-subtiles {2*(w&1), 2*(w&1)+1}, N-tiles 4*(w>>1)..+3 -> 8 WMMAs
// per K step from 2 A + 4 B fragments.
// ---------------------------------------------------------------------------
__global__ __launch_bounds__(256)
void gemm_gnb_kernel(const float* __restrict__ s0, const float* __restrict__ s1,
                     const float* __restrict__ s2, const float* __restrict__ s3,
                     const int* __restrict__ i0, const int* __restrict__ i1,
                     const int* __restrict__ i2, const int* __restrict__ i3,
                     int nseg,
                     const bf16_t* __restrict__ Wt,      // panel-tiled [K/32][256][32]
                     const float* __restrict__ bias,     // [256]
                     const float* __restrict__ rowmask,  // [M] or nullptr
                     float* __restrict__ out)            // [M x 256]
{
    __shared__ bf16_t As[64][APAD];
    __shared__ bf16_t Bs[256][BPAD];

    const float* segs[4] = { s0, s1, s2, s3 };
    const int*   idxs[4] = { i0, i1, i2, i3 };

    const int m0    = blockIdx.x * 64;
    const int tid   = threadIdx.x;
    const int wave  = tid >> 5;
    const int lane  = tid & 31;
    const int half  = lane >> 4;        // lane group 0|1
    const int lr    = lane & 15;
    const int mpair = wave & 1;         // M-subtiles {2*mpair, 2*mpair+1}
    const int nb    = (wave >> 1) << 2; // first of 4 owned N-tiles

    // A-staging geometry: thread -> row (tid>>2), col group (tid&3)*8
    const int arr = tid >> 2;
    const int ac8 = (tid & 3) << 3;
    const unsigned bs0 = (unsigned)(uintptr_t)&Bs[0][0];

    v8f acc[2][4] = {};

    for (int seg = 0; seg < nseg; ++seg) {
        // hoisted per-segment row gather (index loaded once per segment)
        const int* ix  = idxs[seg];
        const int arow = m0 + arr;
        const int srow = ix ? ix[arow] : arow;
        const float* abase = segs[seg] + (size_t)srow * DIM + ac8;
        const bf16_t* pbase = Wt + ((size_t)seg << 8) * 32;   // seg*256 k-cols

        for (int kk = 0; kk < 256; kk += 32) {
            __syncthreads();   // previous iteration's fragment reads done

            // ---- B panel -> LDS via async b128 (contiguous 16KB panel) ---
            {
                const uint64_t pg = (uint64_t)(const void*)(pbase + (size_t)(kk >> 5) * (256 * 32));
                #pragma unroll
                for (int q = 0; q < 4; ++q) {
                    const int i  = tid + (q << 8);          // 0..1023 x 16B
                    const unsigned lds = bs0 + (unsigned)((i >> 2) * (BPAD * 2) + (i & 3) * 16);
                    const uint64_t ga  = pg + (uint64_t)i * 16;
                    asm volatile("global_load_async_to_lds_b128 %0, %1, off"
                                 :: "v"(lds), "v"(ga) : "memory");
                }
            }
            // ---- A tile: 64 rows x 32 cols, f32 -> bf16 ------------------
            {
                const float4 v0 = *(const float4*)(abase + kk);
                const float4 v1 = *(const float4*)(abase + kk + 4);
                As[arr][ac8 + 0] = to_bf16(v0.x);
                As[arr][ac8 + 1] = to_bf16(v0.y);
                As[arr][ac8 + 2] = to_bf16(v0.z);
                As[arr][ac8 + 3] = to_bf16(v0.w);
                As[arr][ac8 + 4] = to_bf16(v1.x);
                As[arr][ac8 + 5] = to_bf16(v1.y);
                As[arr][ac8 + 6] = to_bf16(v1.z);
                As[arr][ac8 + 7] = to_bf16(v1.w);
            }
            asm volatile("s_wait_asynccnt 0x0" ::: "memory");
            __syncthreads();

            // ---- fragments (wave32, 16-bit), contiguous k = 16*half + s in A and B ---------------
            // A 16x32: VGPR j pair kb = 16*half + 2j
            v16bf a[2];
            #pragma unroll
            for (int ms = 0; ms < 2; ++ms) {
                const int am = ((mpair * 2 + ms) << 4) + lr;
                #pragma unroll
                for (int j = 0; j < 8; ++j) {
                    const int kb = (half << 4) + (j << 1);
                    a[ms][2 * j]     = As[am][kb];
                    a[ms][2 * j + 1] = As[am][kb + 1];
                }
            }
            // B 32x16: col = lane&15; VGPR j pair kb = 16*half + 2j
            #pragma unroll
            for (int t = 0; t < 4; ++t) {
                const int bn = ((nb + t) << 4) + lr;
                v16bf b;
                #pragma unroll
                for (int j = 0; j < 8; ++j) {
                    const int kb = (half << 4) + (j << 1);
                    b[2 * j]     = Bs[bn][kb];
                    b[2 * j + 1] = Bs[bn][kb + 1];
                }
                acc[0][t] = __builtin_amdgcn_wmma_f32_16x16x32_bf16(
                                false, a[0], false, b, (short)0, acc[0][t], false, false);
                acc[1][t] = __builtin_amdgcn_wmma_f32_16x16x32_bf16(
                                false, a[1], false, b, (short)0, acc[1][t], false, false);
            }
        }
    }

    // ---- epilogue: bias + relu (+mask). C layout: m = v + 8*half ---------
    #pragma unroll
    for (int t = 0; t < 4; ++t) {
        const int n  = ((nb + t) << 4) + lr;
        const float bv = bias[n];
        #pragma unroll
        for (int ms = 0; ms < 2; ++ms) {
            #pragma unroll
            for (int v = 0; v < 8; ++v) {
                const int m   = ((mpair * 2 + ms) << 4) + (half << 3) + v;
                const int row = m0 + m;
                float val = acc[ms][t][v] + bv;
                val = val > 0.0f ? val : 0.0f;
                if (rowmask) val *= rowmask[row];
                out[(size_t)row * DIM + n] = val;
            }
        }
    }
}

// ---------------------------------------------------------------------------
// rho_e_v: per-node mean of incoming e_prime rows. One wave per node scans
// its graph's edge block in order (deterministic). Lane owns 8 columns.
// ---------------------------------------------------------------------------
__global__ __launch_bounds__(256)
void rho_e_v_kernel(const float* __restrict__ e_prime, const int* __restrict__ r,
                    const int* __restrict__ edge_offsets,
                    float* __restrict__ e_agg_node, int max_order)
{
    const int wave = threadIdx.x >> 5;
    const int lane = threadIdx.x & 31;
    const int node = blockIdx.x * 8 + wave;
    const int g    = node / max_order;
    const int eb   = edge_offsets[g];
    const int ee   = edge_offsets[g + 1];

    float a[8] = {0, 0, 0, 0, 0, 0, 0, 0};
    int cnt = 0;

    for (int e0 = eb; e0 < ee; e0 += 32) {
        const int my_r = r[e0 + lane];
        #pragma unroll 1
        for (int i = 0; i < 32; ++i) {
            const int rr = __builtin_amdgcn_readlane(my_r, i);
            if (rr == node) {
                const float4* row = (const float4*)(e_prime + (size_t)(e0 + i) * DIM);
                const float4 p0 = row[2 * lane];
                const float4 p1 = row[2 * lane + 1];
                a[0] += p0.x; a[1] += p0.y; a[2] += p0.z; a[3] += p0.w;
                a[4] += p1.x; a[5] += p1.y; a[6] += p1.z; a[7] += p1.w;
                ++cnt;
            }
        }
    }
    const float inv = (cnt > 0) ? (1.0f / (float)cnt) : 0.0f;
    float* o = e_agg_node + (size_t)node * DIM + lane * 8;
    float4 q0 = { a[0] * inv, a[1] * inv, a[2] * inv, a[3] * inv };
    float4 q1 = { a[4] * inv, a[5] * inv, a[6] * inv, a[7] * inv };
    *(float4*)(o)     = q0;
    *(float4*)(o + 4) = q1;
}

// ---------------------------------------------------------------------------
// rho_e_u: per-graph mean of e_prime over its contiguous edge block.
// ---------------------------------------------------------------------------
__global__ __launch_bounds__(256)
void rho_e_u_kernel(const float* __restrict__ e_prime,
                    const int* __restrict__ edge_offsets,
                    float* __restrict__ e_agg_g)
{
    const int g = blockIdx.x, col = threadIdx.x;
    const int eb = edge_offsets[g], ee = edge_offsets[g + 1];
    float s = 0.0f;
    for (int e = eb; e < ee; ++e) s += e_prime[(size_t)e * DIM + col];
    e_agg_g[(size_t)g * DIM + col] = s / (float)(ee - eb);
}

// ---------------------------------------------------------------------------
// rho_v_u: per-graph sum of V_prime over padded nodes / graph_order.
// ---------------------------------------------------------------------------
__global__ __launch_bounds__(256)
void rho_v_u_kernel(const float* __restrict__ V_prime,
                    const int* __restrict__ graph_orders,
                    float* __restrict__ v_agg, int max_order)
{
    const int g = blockIdx.x, col = threadIdx.x;
    float s = 0.0f;
    for (int i = 0; i < max_order; ++i)
        s += V_prime[((size_t)g * max_order + i) * DIM + col];
    v_agg[(size_t)g * DIM + col] = s / (float)graph_orders[g];
}

// ---------------------------------------------------------------------------
extern "C" void kernel_launch(void* const* d_in, const int* in_sizes, int n_in,
                              void* d_out, int out_size, void* d_ws, size_t ws_size,
                              hipStream_t stream)
{
    // setup_inputs() order:
    const float* u            = (const float*)d_in[0];
    const float* V            = (const float*)d_in[1];
    const float* e            = (const float*)d_in[2];
    const int*   s            = (const int*)d_in[3];
    const int*   r            = (const int*)d_in[4];
    const int*   graph_orders = (const int*)d_in[5];
    const float* node_mask    = (const float*)d_in[6];
    const int*   edge_offsets = (const int*)d_in[7];
    const float* u_by_nodes   = (const float*)d_in[8];
    const float* u_by_edges   = (const float*)d_in[9];
    const float* W_e          = (const float*)d_in[10];
    const float* b_e          = (const float*)d_in[11];
    const float* W_v          = (const float*)d_in[12];
    const float* b_v          = (const float*)d_in[13];
    const float* W_u          = (const float*)d_in[14];
    const float* b_u          = (const float*)d_in[15];

    const int n_graphs  = in_sizes[0] / DIM;   // 64
    const int n_nodes   = in_sizes[1] / DIM;   // 8192
    const int n_edges   = in_sizes[2] / DIM;   // 131072
    const int max_order = n_nodes / n_graphs;  // 128
    const int Ke = 4 * DIM;                    // 1024
    const int Kv = 3 * DIM;                    // 768
    const int Ku = 3 * DIM;                    // 768

    // d_out = (u_prime[1,G,256], V_prime[N,256], e_prime[E,256]) flat
    float* out_u = (float*)d_out;
    float* out_V = out_u + (size_t)n_graphs * DIM;
    float* out_e = out_V + (size_t)n_nodes * DIM;

    // workspace carve-up (256B aligned)
    char* ws = (char*)d_ws;
    size_t ofs = 0;
    auto carve = [&](size_t bytes) { void* p = ws + ofs; ofs += (bytes + 255) & ~(size_t)255; return p; };
    bf16_t* WtE        = (bf16_t*)carve((size_t)DIM * Ke * sizeof(bf16_t));
    bf16_t* WtV        = (bf16_t*)carve((size_t)DIM * Kv * sizeof(bf16_t));
    bf16_t* WtU        = (bf16_t*)carve((size_t)DIM * Ku * sizeof(bf16_t));
    float*  e_agg_node = (float*)carve((size_t)n_nodes * DIM * sizeof(float));
    float*  e_agg_g    = (float*)carve((size_t)n_graphs * DIM * sizeof(float));
    float*  v_agg      = (float*)carve((size_t)n_graphs * DIM * sizeof(float));
    (void)ws_size;

    // 1) weights -> bf16, K-panel tiled
    wconv_kernel<<<(Ke * DIM + 255) / 256, 256, 0, stream>>>(W_e, WtE, Ke);
    wconv_kernel<<<(Kv * DIM + 255) / 256, 256, 0, stream>>>(W_v, WtV, Kv);
    wconv_kernel<<<(Ku * DIM + 255) / 256, 256, 0, stream>>>(W_u, WtU, Ku);

    // 2) f_e: e_prime = relu([e | V[s] | V[r] | u_by_edges] @ W_e + b_e)
    gemm_gnb_kernel<<<n_edges / 64, 256, 0, stream>>>(
        e, V, V, u_by_edges,
        nullptr, s, r, nullptr,
        4, WtE, b_e, nullptr, out_e);

    // 3) aggregations over e_prime (deterministic)
    rho_e_v_kernel<<<n_nodes / 8, 256, 0, stream>>>(out_e, r, edge_offsets,
                                                    e_agg_node, max_order);
    rho_e_u_kernel<<<n_graphs, 256, 0, stream>>>(out_e, edge_offsets, e_agg_g);

    // 4) f_v: V_prime = relu([e_agg_node | V | u_by_nodes] @ W_v + b_v) * mask
    gemm_gnb_kernel<<<n_nodes / 64, 256, 0, stream>>>(
        e_agg_node, V, u_by_nodes, nullptr,
        nullptr, nullptr, nullptr, nullptr,
        3, WtV, b_v, node_mask, out_V);

    // 5) rho_v_u
    rho_v_u_kernel<<<n_graphs, 256, 0, stream>>>(out_V, graph_orders, v_agg, max_order);

    // 6) f_u: u_prime = relu([e_agg_g | v_agg | u] @ W_u + b_u)
    gemm_gnb_kernel<<<n_graphs / 64, 256, 0, stream>>>(
        e_agg_g, v_agg, u, nullptr,
        nullptr, nullptr, nullptr, nullptr,
        3, WtU, b_u, nullptr, out_u);
}
